// RPN_73091753443935
// MI455X (gfx1250) — matrix operand order repaired and checked by reading, not compiled, not run on hardware
//
#include <hip/hip_runtime.h>
#include <hip/hip_bf16.h>

typedef __attribute__((ext_vector_type(16))) _Float16 v16h;
typedef __attribute__((ext_vector_type(8)))  _Float16 v8h;
typedef __attribute__((ext_vector_type(8)))  float    v8f;

#define HF 128
#define WF 128
#define CIN 128
#define LN1000 6.907755278982137f

// d_ws layout (bytes)
#define FEATT_OFF 0u           // 8*128*128*128 f16 = 33,554,432 B
#define FEATR_OFF 33554432u    // 33,554,432 B
#define BP1_OFF   67108864u    // 9*4*8*32*16 f16 = 294,912 B
#define BP2_OFF   67403776u    // 4*3*32*16 f16 = 12,288 B

// out regions (float elements)
#define BOX_OFF   1179648
#define PROP_OFF  5898240

// ---------------------------------------------------------------------------
// NCHW f32 -> NHWC f16 via LDS transpose (coalesced read + coalesced write)
// ---------------------------------------------------------------------------
__global__ __launch_bounds__(1024) void pack_feat_nhwc_f16(
    const float* __restrict__ in, _Float16* __restrict__ out)
{
    __shared__ _Float16 tile[32][33];
    const int tx = threadIdx.x, ty = threadIdx.y;
    const int cBase = (blockIdx.x & 3) << 5;
    const int xBase = (blockIdx.x >> 2) << 5;
    const int by = blockIdx.y;            // b*128 + y
    const int b = by >> 7, y = by & 127;
    tile[ty][tx] = (_Float16)in[(((b * CIN + cBase + ty) * HF + y) * WF) + xBase + tx];
    __syncthreads();
    out[(((b * HF + y) * WF + xBase + ty) * CIN) + cBase + tx] = tile[tx][ty];
}

// ---------------------------------------------------------------------------
// conv1 weights (O=128,I=128,3,3) f32 -> WMMA B-fragment-packed f16
// frag index: (((kpos*4 + ks)*8 + nt)*32 + lane)*16 + h
// lane<16: h[k]=B[K=ks*32+k][n], lane>=16: h[k]=B[K=ks*32+16+k][n]; n=nt*16+lane%16
// ---------------------------------------------------------------------------
__global__ void pack_w1(const float* __restrict__ w, _Float16* __restrict__ bp)
{
    const int idx  = blockIdx.x * 256 + threadIdx.x;   // 147456 total
    const int h    = idx & 15;
    const int lane = (idx >> 4) & 31;
    const int nt   = (idx >> 9) & 7;
    const int ks   = (idx >> 12) & 3;
    const int kpos = idx >> 14;                        // 0..8 (dy*3+dx)
    const int c = ks * 32 + ((lane < 16) ? h : 16 + h);
    const int n = nt * 16 + (lane & 15);
    bp[idx] = (_Float16)w[(n * CIN + c) * 9 + kpos];
}

// cls (9,128) + reg (36,128) f32 -> packed f16, N padded to 48 (3 tiles)
__global__ void pack_w2(const float* __restrict__ clsw, const float* __restrict__ regw,
                        _Float16* __restrict__ bp)
{
    const int idx  = blockIdx.x * 256 + threadIdx.x;   // 6144 total
    const int h    = idx & 15;
    const int lane = (idx >> 4) & 31;
    const int rest = idx >> 9;                         // ks*3 + nt
    const int nt   = rest % 3;
    const int ks   = rest / 3;
    const int c = ks * 32 + ((lane < 16) ? h : 16 + h);
    const int n = nt * 16 + (lane & 15);
    float v = 0.0f;
    if (n < 9)       v = clsw[n * CIN + c];
    else if (n < 45) v = regw[(n - 9) * CIN + c];
    bp[idx] = (_Float16)v;
}

// ---------------------------------------------------------------------------
// conv1 3x3 + bias + ReLU as implicit GEMM, f16 WMMA, f32 accumulate.
// B staged per-kpos into LDS via async copies (ASYNCcnt path); per K-step all
// 8 B fragments are batch-loaded (ds_load_b128 x16, one wait) so the 8 WMMAs
// issue back-to-back. One wave = 16-pixel x 128-channel tile.
// ---------------------------------------------------------------------------
__global__ __launch_bounds__(256) void conv3x3_wmma(
    const _Float16* __restrict__ featT, const _Float16* __restrict__ Bp,
    const float* __restrict__ bias, _Float16* __restrict__ featR)
{
    __shared__ _Float16 Bs[16384];                     // 32 KB: one kpos slice
    const int lid  = threadIdx.x;
    const int lane = lid & 31;
    const int wave = lid >> 5;
    const int tile = blockIdx.x * 8 + wave;            // 8192 tiles
    const int m    = lane & 15;
    const int hiK  = lane >> 4;
    const int x0 = (tile & 7) << 4;
    const int y  = (tile >> 3) & 127;
    const int b  = tile >> 10;

    // LDS byte address of Bs (low 32 bits of the generic LDS-aperture address)
    const unsigned lds_base = (unsigned)(uintptr_t)(void*)Bs;

    v8f acc[8];
    #pragma unroll
    for (int nt = 0; nt < 8; ++nt) {
        const float bv = bias[nt * 16 + m];
        #pragma unroll
        for (int r = 0; r < 8; ++r) acc[nt][r] = bv;
    }

    const v8h Z8 = {(_Float16)0.f,(_Float16)0.f,(_Float16)0.f,(_Float16)0.f,
                    (_Float16)0.f,(_Float16)0.f,(_Float16)0.f,(_Float16)0.f};
    const int px = x0 + m;
    const v16h* __restrict__ bls = (const v16h*)Bs;

    #pragma unroll 1
    for (int kpos = 0; kpos < 9; ++kpos) {
        // ---- stage this kpos's 32KB B slice into LDS (async copy path) ----
        __syncthreads();   // previous slice fully consumed
        {
            const unsigned long long gbase =
                (unsigned long long)(const void*)(Bp + kpos * 16384);
            #pragma unroll
            for (int i = 0; i < 8; ++i) {
                unsigned ldsa = lds_base + (unsigned)(i * 4096 + lid * 16);
                unsigned goff = (unsigned)(i * 4096 + lid * 16);
                asm volatile("global_load_async_to_lds_b128 %0, %1, %2"
                             :: "v"(ldsa), "v"(goff), "s"(gbase) : "memory");
            }
            asm volatile("s_wait_asynccnt 0x0" ::: "memory");
        }
        __syncthreads();   // all waves' slices landed

        const int dy = kpos / 3, dx = kpos % 3;
        const int yy = y + dy - 1;
        const int xx = px + dx - 1;
        const bool valid = ((unsigned)yy < 128u) && ((unsigned)xx < 128u);
        const _Float16* arow = featT + ((((b << 7) + yy) << 7) + xx) * CIN;
        #pragma unroll
        for (int ks = 0; ks < 4; ++ks) {
            const int cb = ks * 32 + hiK * 16;
            v8h lo = valid ? *(const v8h*)(arow + cb)      : Z8;
            v8h hi = valid ? *(const v8h*)(arow + cb + 8) : Z8;
            v16h A = __builtin_shufflevector(lo, hi, 0,1,2,3,4,5,6,7,8,9,10,11,12,13,14,15);
            // batch-load all 8 B fragments for this K-step, then 8 WMMAs
            v16h Bf[8];
            #pragma unroll
            for (int nt = 0; nt < 8; ++nt)
                Bf[nt] = bls[(ks * 8 + nt) * 32 + lane];
            #pragma unroll
            for (int nt = 0; nt < 8; ++nt)
                acc[nt] = __builtin_amdgcn_wmma_f32_16x16x32_f16(
                    false, A, false, Bf[nt], (short)0, acc[nt], false, false);
        }
    }

    // ReLU + store NHWC f16 (D layout: c[r] -> pixel r + hiK*8, channel nt*16+m)
    _Float16* orow = featR + ((((b << 7) + y) << 7) + x0) * CIN;
    #pragma unroll
    for (int nt = 0; nt < 8; ++nt) {
        #pragma unroll
        for (int r = 0; r < 8; ++r) {
            float v = acc[nt][r];
            v = v > 0.0f ? v : 0.0f;
            orow[(r + hiK * 8) * CIN + nt * 16 + m] = (_Float16)v;
        }
    }
}

// ---------------------------------------------------------------------------
// Heads: 1x1 convs (cls 9 + reg 36, padded to N=48) as one WMMA GEMM, K=128.
// B (12KB) staged once into LDS via async copies; fragments batch-loaded.
// ---------------------------------------------------------------------------
__global__ __launch_bounds__(256) void heads_wmma(
    const _Float16* __restrict__ featR, const _Float16* __restrict__ Bp2,
    const float* __restrict__ cls_b, const float* __restrict__ reg_b,
    float* __restrict__ out)
{
    __shared__ _Float16 Bs2[6144];                     // 12 KB
    const int lid  = threadIdx.x;
    const int lane = lid & 31;
    const int wave = lid >> 5;
    const int tile = blockIdx.x * 8 + wave;
    const int m    = lane & 15;
    const int hiK  = lane >> 4;
    const int x0 = (tile & 7) << 4;
    const int y  = (tile >> 3) & 127;
    const int b  = tile >> 10;

    {
        const unsigned lds_base = (unsigned)(uintptr_t)(void*)Bs2;
        const unsigned long long gbase = (unsigned long long)(const void*)Bp2;
        #pragma unroll
        for (int i = 0; i < 3; ++i) {
            unsigned ldsa = lds_base + (unsigned)(i * 4096 + lid * 16);
            unsigned goff = (unsigned)(i * 4096 + lid * 16);
            asm volatile("global_load_async_to_lds_b128 %0, %1, %2"
                         :: "v"(ldsa), "v"(goff), "s"(gbase) : "memory");
        }
        asm volatile("s_wait_asynccnt 0x0" ::: "memory");
    }
    __syncthreads();

    v8f acc[3];
    #pragma unroll
    for (int nt = 0; nt < 3; ++nt) {
        const int n = nt * 16 + m;
        const float bv = (n < 9) ? cls_b[n] : ((n < 45) ? reg_b[n - 9] : 0.0f);
        #pragma unroll
        for (int r = 0; r < 8; ++r) acc[nt][r] = bv;
    }

    const _Float16* arow = featR + ((((b << 7) + y) << 7) + x0 + m) * CIN;
    const v16h* __restrict__ bls = (const v16h*)Bs2;
    #pragma unroll
    for (int ks = 0; ks < 4; ++ks) {
        const int cb = ks * 32 + hiK * 16;
        v8h lo = *(const v8h*)(arow + cb);
        v8h hi = *(const v8h*)(arow + cb + 8);
        v16h A = __builtin_shufflevector(lo, hi, 0,1,2,3,4,5,6,7,8,9,10,11,12,13,14,15);
        v16h Bf[3];
        #pragma unroll
        for (int nt = 0; nt < 3; ++nt)
            Bf[nt] = bls[(ks * 3 + nt) * 32 + lane];
        #pragma unroll
        for (int nt = 0; nt < 3; ++nt)
            acc[nt] = __builtin_amdgcn_wmma_f32_16x16x32_f16(
                false, A, false, Bf[nt], (short)0, acc[nt], false, false);
    }

    const int pixBase = (((b << 7) + y) << 7) + x0;
    #pragma unroll
    for (int nt = 0; nt < 3; ++nt) {
        const int n = nt * 16 + m;
        #pragma unroll
        for (int r = 0; r < 8; ++r) {
            const int pix = pixBase + r + hiK * 8;
            if (n < 9)       out[pix * 9 + n] = acc[nt][r];
            else if (n < 45) out[BOX_OFF + pix * 36 + (n - 9)] = acc[nt][r];
        }
    }
}

// ---------------------------------------------------------------------------
// Anchors + box transform (elementwise, anchors recomputed on the fly)
// ---------------------------------------------------------------------------
__global__ void proposals_kernel(const float* __restrict__ box, float* __restrict__ prop)
{
    const int idx = blockIdx.x * 256 + threadIdx.x;    // 1,179,648 total
    const int k = idx % 147456;
    const int a = k % 9;
    const int cell = k / 9;
    const int x = cell & 127;
    const int yy = (cell >> 7) & 127;

    const float4 t = ((const float4*)box)[idx];

    const float scl_tab[3] = {128.0f, 256.0f, 512.0f};
    const float hr_tab[3]  = {0.70710678118654752f, 1.0f, 1.41421356237309505f}; // sqrt(ar)
    const int s = a % 3, i = a / 3;
    const float hr = hr_tab[i];
    const float wr = 1.0f / hr;
    const float wsc = wr * scl_tab[s];
    const float hsc = hr * scl_tab[s];
    const float bx2 = rintf(0.5f * wsc), bx0 = rintf(-0.5f * wsc);
    const float by2 = rintf(0.5f * hsc), by0 = rintf(-0.5f * hsc);
    const float aw = bx2 - bx0, ah = by2 - by0;
    const float sx = (float)(x * 8), sy = (float)(yy * 8);

    float4 o;
    o.x = t.x * (aw + sx + bx0);
    o.y = t.z * (ah + sy + by0);
    o.z = expf(fminf(t.y, LN1000)) * aw;
    o.w = expf(fminf(t.w, LN1000)) * ah;
    ((float4*)prop)[idx] = o;
}

// ---------------------------------------------------------------------------
extern "C" void kernel_launch(void* const* d_in, const int* in_sizes, int n_in,
                              void* d_out, int out_size, void* d_ws, size_t ws_size,
                              hipStream_t stream)
{
    // setup_inputs order: image, feat, conv1_w, conv1_b, reg_w, reg_b, cls_w, cls_b
    const float* feat = (const float*)d_in[1];
    const float* c1w  = (const float*)d_in[2];
    const float* c1b  = (const float*)d_in[3];
    const float* regw = (const float*)d_in[4];
    const float* regb = (const float*)d_in[5];
    const float* clsw = (const float*)d_in[6];
    const float* clsb = (const float*)d_in[7];

    char* ws = (char*)d_ws;
    _Float16* featT = (_Float16*)(ws + FEATT_OFF);
    _Float16* featR = (_Float16*)(ws + FEATR_OFF);
    _Float16* Bp1   = (_Float16*)(ws + BP1_OFF);
    _Float16* Bp2   = (_Float16*)(ws + BP2_OFF);
    float* out = (float*)d_out;

    pack_feat_nhwc_f16<<<dim3(16, 1024), dim3(32, 32), 0, stream>>>(feat, featT);
    pack_w1<<<576, 256, 0, stream>>>(c1w, Bp1);
    pack_w2<<<24, 256, 0, stream>>>(clsw, regw, Bp2);
    conv3x3_wmma<<<1024, 256, 0, stream>>>(featT, Bp1, c1b, featR);
    heads_wmma<<<1024, 256, 0, stream>>>(featR, Bp2, clsb, regb, out);
    proposals_kernel<<<4608, 256, 0, stream>>>(out + BOX_OFF, out + PROP_OFF);
}
